// VAE_40853728919732
// MI455X (gfx1250) — matrix operand order repaired and checked by reading, not compiled, not run on hardware
//
#include <hip/hip_runtime.h>

// ---------------------------------------------------------------------------
// Types for CDNA5 WMMA (wave32, 16x16x32 bf16 -> f32 accumulate)
// ---------------------------------------------------------------------------
typedef __attribute__((ext_vector_type(16))) __bf16 v16bf;
typedef __attribute__((ext_vector_type(8)))  __bf16 v8bf;
typedef __attribute__((ext_vector_type(8)))  float  v8f;

#define USE_ASYNC_LDS 1

static __device__ inline unsigned short f2bf_bits(float f) {
    unsigned u = __builtin_bit_cast(unsigned, f);
    unsigned r = u + 0x7FFFu + ((u >> 16) & 1u);   // round-to-nearest-even
    return (unsigned short)(r >> 16);
}

// A-fragment style load: 16 contiguous elements
static __device__ inline v16bf ld_frag_split(const unsigned short* p) {
    v8bf lo = *(const v8bf*)(p);
    v8bf hi = *(const v8bf*)(p + 8);
    return __builtin_shufflevector(lo, hi, 0,1,2,3,4,5,6,7,8,9,10,11,12,13,14,15);
}
// B-fragment style load: 16 contiguous elements
static __device__ inline v16bf ld_frag_contig(const unsigned short* p) {
    v8bf lo = *(const v8bf*)(p);
    v8bf hi = *(const v8bf*)(p + 8);
    return __builtin_shufflevector(lo, hi, 0,1,2,3,4,5,6,7,8,9,10,11,12,13,14,15);
}

static __device__ inline v8f wmma_bf16(v16bf a, v16bf b, v8f c) {
    return __builtin_amdgcn_wmma_f32_16x16x32_bf16(
        /*neg_a=*/false, a, /*neg_b=*/false, b,
        /*c_mod=*/(short)0, c, /*reuse_a=*/false, /*reuse_b=*/false);
}

#if USE_ASYNC_LDS
// CDNA5 async copy: memory -> LDS, 16B per lane, tracked by ASYNCcnt.
// VDST = per-lane LDS byte address (low 32 bits of generic pointer to LDS).
static __device__ inline void async_ld_b128(const unsigned short* gptr,
                                            unsigned short* lptr) {
    unsigned lds_off = (unsigned)(unsigned long long)(void*)lptr;
    asm volatile("global_load_async_to_lds_b128 %0, %1, off"
                 :: "v"(lds_off), "v"((unsigned long long)gptr)
                 : "memory");
}
static __device__ inline void wait_async0() {
    asm volatile("s_wait_asynccnt 0x0" ::: "memory");
}
#endif

// ---------------------------------------------------------------------------
// fp32 -> bf16 conversion
// ---------------------------------------------------------------------------
__global__ void f2bf_kernel(const float* __restrict__ in,
                            unsigned short* __restrict__ out, int n) {
    int i = blockIdx.x * blockDim.x + threadIdx.x;
    if (i < n) out[i] = f2bf_bits(in[i]);
}

// ---------------------------------------------------------------------------
// Tiled WMMA GEMM: out[M,N] = act( A[M,K](bf16) @ W[N,K]^T(bf16) + bias
//                                  (+ resid[M,N]) )
// Block: 256 threads (8 waves). Block tile 128x128, wave tile 32x64.
// Double-buffered LDS, async global->LDS copies, one barrier per K-step.
// ---------------------------------------------------------------------------
__global__ __launch_bounds__(256) void gemm_wmma(
    const unsigned short* __restrict__ A,   // M x K, bf16
    const unsigned short* __restrict__ W,   // N x K, bf16 (row-major)
    const float* __restrict__ bias,         // N or null
    const float* __restrict__ resid,        // M x N or null
    float* __restrict__ outF,               // M x N or null
    unsigned short* __restrict__ outB,      // M x N bf16 or null
    int M, int N, int K, int relu)
{
    __shared__ unsigned short sA[2][128 * 48]; // 48-ushort stride (96B, 16B aligned)
    __shared__ unsigned short sB[2][128 * 48];

    const int tid  = threadIdx.x;
    const int lane = tid & 31;
    const int wv   = tid >> 5;
    const int wm   = wv & 3;        // 4 wave-rows of 32
    const int wn   = wv >> 2;       // 2 wave-cols of 64
    const int m0   = blockIdx.y * 128;
    const int n0   = blockIdx.x * 128;

    v8f acc[2][4] = {};

    const int trow  = tid >> 1;          // 0..127 (row for both A and B tiles)
    const int thalf = (tid & 1) * 16;    // 0 / 16 elements within 32-wide K slab

    const int lr = lane & 15;
    const int k0 = (lane < 16) ? 0 : 16;  // A-frag contig-run start
    const int kc = (lane < 16) ? 0 : 16;  // B-frag contig-run start
    const int mh = (lane >> 4) * 8;       // C-frag row offset

    auto load_tile = [&](int buf, int kb) {
        const unsigned short* ga = A + (size_t)(m0 + trow) * K + kb + thalf;
        unsigned short* la = &sA[buf][trow * 48 + thalf];
        const unsigned short* gw = W + (size_t)(n0 + trow) * K + kb + thalf;
        unsigned short* lb = &sB[buf][trow * 48 + thalf];
#if USE_ASYNC_LDS
        async_ld_b128(ga,     la);
        async_ld_b128(ga + 8, la + 8);
        async_ld_b128(gw,     lb);
        async_ld_b128(gw + 8, lb + 8);
#else
        const uint4* g4a = (const uint4*)ga;
        *(uint4*)(la)     = g4a[0];
        *(uint4*)(la + 8) = g4a[1];
        const uint4* g4b = (const uint4*)gw;
        *(uint4*)(lb)     = g4b[0];
        *(uint4*)(lb + 8) = g4b[1];
#endif
    };

    // Prologue: stage first K-slab
    load_tile(0, 0);
#if USE_ASYNC_LDS
    wait_async0();
#endif
    __syncthreads();

    const int nk = K >> 5;
    for (int s = 0; s < nk; s++) {
        const int cur = s & 1;
        if (s + 1 < nk) load_tile(cur ^ 1, (s + 1) << 5);

        v16bf a0 = ld_frag_split(&sA[cur][(wm * 32      + lr) * 48 + k0]);
        v16bf a1 = ld_frag_split(&sA[cur][(wm * 32 + 16 + lr) * 48 + k0]);
        v16bf bf0 = ld_frag_contig(&sB[cur][(wn * 64      + lr) * 48 + kc]);
        v16bf bf1 = ld_frag_contig(&sB[cur][(wn * 64 + 16 + lr) * 48 + kc]);
        v16bf bf2 = ld_frag_contig(&sB[cur][(wn * 64 + 32 + lr) * 48 + kc]);
        v16bf bf3 = ld_frag_contig(&sB[cur][(wn * 64 + 48 + lr) * 48 + kc]);

        acc[0][0] = wmma_bf16(a0, bf0, acc[0][0]);
        acc[0][1] = wmma_bf16(a0, bf1, acc[0][1]);
        acc[0][2] = wmma_bf16(a0, bf2, acc[0][2]);
        acc[0][3] = wmma_bf16(a0, bf3, acc[0][3]);
        acc[1][0] = wmma_bf16(a1, bf0, acc[1][0]);
        acc[1][1] = wmma_bf16(a1, bf1, acc[1][1]);
        acc[1][2] = wmma_bf16(a1, bf2, acc[1][2]);
        acc[1][3] = wmma_bf16(a1, bf3, acc[1][3]);

#if USE_ASYNC_LDS
        wait_async0();   // this wave's async writes to the other buffer are done
#endif
        __syncthreads(); // all waves done reading cur / writing cur^1
    }

    // Epilogue: bias, residual, relu, fp32/bf16 stores
    #pragma unroll
    for (int i = 0; i < 2; i++) {
        #pragma unroll
        for (int j = 0; j < 4; j++) {
            int n = n0 + wn * 64 + j * 16 + lr;
            float bv = bias ? bias[n] : 0.0f;
            #pragma unroll
            for (int v = 0; v < 8; v++) {
                int m = m0 + wm * 32 + i * 16 + mh + v;
                float val = acc[i][j][v] + bv;
                if (resid) val += resid[(size_t)m * N + n];
                if (relu)  val  = fmaxf(val, 0.0f);
                if (outF)  outF[(size_t)m * N + n] = val;
                if (outB)  outB[(size_t)m * N + n] = f2bf_bits(val);
            }
        }
    }
}

// ---------------------------------------------------------------------------
// Flash-style attention, all-WMMA.
// qkv: [B*S, 3*D] bf16 (q | k | v). o: [B*S, D] bf16 (heads concatenated).
// Grid: (S/64, H, B), block 128 (4 waves); wave handles 16 query rows.
// ---------------------------------------------------------------------------
__global__ __launch_bounds__(128) void attn_wmma(
    const unsigned short* __restrict__ qkv,
    unsigned short* __restrict__ o,
    int S, int D, int H)
{
    const int b  = blockIdx.z, h = blockIdx.y, qt = blockIdx.x;
    const int tid = threadIdx.x, lane = tid & 31, wv = tid >> 5;
    const int dh = D / H;            // 64
    const int row3 = 3 * D;

    __shared__ unsigned short sVt[64 * 72];  // V chunk, transposed [d][s_local]
    __shared__ unsigned short sP[64 * 72];   // probabilities, [q_local][s_local]

    const int lr = lane & 15;
    const int hi = lane >> 4;
    const int k0 = hi ? 16 : 0;     // A-frag contig-run start
    const int kc = hi ? 16 : 0;     // B-frag contig-run start
    const int mh = hi * 8;          // C-frag row offset

    // Preload Q fragments (rows qt*64 + wv*16 + lr, dh=64 -> 2 K-steps)
    const size_t qbase = ((size_t)b * S + qt * 64 + wv * 16 + lr) * row3 + h * dh;
    v16bf qf[2];
    #pragma unroll
    for (int ks = 0; ks < 2; ks++)
        qf[ks] = ld_frag_split(qkv + qbase + ks * 32 + k0);

    float mrun[8], ssum[8];
    v8f oacc[4] = {};
    #pragma unroll
    for (int v = 0; v < 8; v++) { mrun[v] = -1e30f; ssum[v] = 0.0f; }

    for (int c = 0; c < S; c += 64) {
        // Cooperatively stage V chunk transposed into LDS
        for (int idx = tid; idx < 64 * 64; idx += 128) {
            int sl = idx >> 6, d = idx & 63;
            sVt[d * 72 + sl] =
                qkv[((size_t)b * S + c + sl) * row3 + 2 * D + h * dh + d];
        }
        __syncthreads();

        // Scores: S_w(16x64) = Qw(16x64) @ Kc(64x64)^T — K frags from global (L2-hot)
        v8f sacc[4] = {};
        #pragma unroll
        for (int ks = 0; ks < 2; ks++) {
            #pragma unroll
            for (int j = 0; j < 4; j++) {
                const size_t krow =
                    ((size_t)b * S + c + j * 16 + lr) * row3 + D + h * dh + ks * 32 + kc;
                v16bf kf = ld_frag_contig(qkv + krow);
                sacc[j] = wmma_bf16(qf[ks], kf, sacc[j]);
            }
        }

        // Online softmax (scale = 1/sqrt(64) = 0.125)
        #pragma unroll
        for (int v = 0; v < 8; v++) {
            float cmax = -1e30f;
            #pragma unroll
            for (int j = 0; j < 4; j++) cmax = fmaxf(cmax, sacc[j][v] * 0.125f);
            for (int off = 8; off; off >>= 1)
                cmax = fmaxf(cmax, __shfl_xor(cmax, off, 32));
            float mn = fmaxf(mrun[v], cmax);
            float al = expf(mrun[v] - mn);
            float rs = 0.0f;
            #pragma unroll
            for (int j = 0; j < 4; j++) {
                float p = expf(sacc[j][v] * 0.125f - mn);
                sacc[j][v] = p;
                rs += p;
            }
            for (int off = 8; off; off >>= 1) rs += __shfl_xor(rs, off, 32);
            ssum[v] = ssum[v] * al + rs;
            mrun[v] = mn;
            #pragma unroll
            for (int j = 0; j < 4; j++) oacc[j][v] *= al;
        }

        // P: accumulator layout -> LDS -> A-fragment layout (intra-wave, LDS in-order)
        #pragma unroll
        for (int j = 0; j < 4; j++)
            #pragma unroll
            for (int v = 0; v < 8; v++)
                sP[(wv * 16 + mh + v) * 72 + j * 16 + lr] = f2bf_bits(sacc[j][v]);

        // O += P(16x64) @ V(64x64)
        #pragma unroll
        for (int ks = 0; ks < 2; ks++) {
            v16bf pf = ld_frag_split(&sP[(wv * 16 + lr) * 72 + ks * 32 + k0]);
            #pragma unroll
            for (int j = 0; j < 4; j++) {
                v16bf vf = ld_frag_contig(&sVt[(j * 16 + lr) * 72 + ks * 32 + kc]);
                oacc[j] = wmma_bf16(pf, vf, oacc[j]);
            }
        }
        __syncthreads();
    }

    // Normalize and store (bf16, heads concatenated)
    #pragma unroll
    for (int j = 0; j < 4; j++)
        #pragma unroll
        for (int v = 0; v < 8; v++) {
            int m = qt * 64 + wv * 16 + mh + v;
            float val = oacc[j][v] / ssum[v];
            o[((size_t)b * S + m) * D + h * dh + j * 16 + lr] = f2bf_bits(val);
        }
}

// ---------------------------------------------------------------------------
// LayerNorm over D=512, one row per block; writes fp32 + bf16 copies.
// ---------------------------------------------------------------------------
__global__ __launch_bounds__(128) void ln_kernel(
    const float* __restrict__ in, const float* __restrict__ g,
    const float* __restrict__ bta, float* __restrict__ outF,
    unsigned short* __restrict__ outB, int Dm)
{
    const int row = blockIdx.x;
    const float* x = in + (size_t)row * Dm;
    __shared__ float red[4];
    const int tid = threadIdx.x, lane = tid & 31, wv = tid >> 5;

    float s = 0.0f;
    for (int i = tid; i < Dm; i += 128) s += x[i];
    for (int off = 16; off; off >>= 1) s += __shfl_xor(s, off, 32);
    if (lane == 0) red[wv] = s;
    __syncthreads();
    float mean = (red[0] + red[1] + red[2] + red[3]) / (float)Dm;
    __syncthreads();

    float vs = 0.0f;
    for (int i = tid; i < Dm; i += 128) { float d = x[i] - mean; vs += d * d; }
    for (int off = 16; off; off >>= 1) vs += __shfl_xor(vs, off, 32);
    if (lane == 0) red[wv] = vs;
    __syncthreads();
    float var = (red[0] + red[1] + red[2] + red[3]) / (float)Dm;
    float inv = rsqrtf(var + 1e-5f);

    for (int i = tid; i < Dm; i += 128) {
        float yv = (x[i] - mean) * inv * g[i] + bta[i];
        outF[(size_t)row * Dm + i] = yv;
        outB[(size_t)row * Dm + i] = f2bf_bits(yv);
    }
}

// ---------------------------------------------------------------------------
// Small fp32 GEMM (head, B=64 rows): out = act(A @ W^T + bias)
// act: 0 none, 1 relu, 2 exp
// ---------------------------------------------------------------------------
__global__ void small_gemm(const float* __restrict__ A, int lda,
                           const float* __restrict__ W,
                           const float* __restrict__ bias,
                           float* __restrict__ out,
                           int M, int N, int K, int act)
{
    int idx = blockIdx.x * blockDim.x + threadIdx.x;
    if (idx >= M * N) return;
    int m = idx / N, n = idx % N;
    const float* a = A + (size_t)m * lda;
    const float* w = W + (size_t)n * K;
    float s = bias ? bias[n] : 0.0f;
    for (int k = 0; k < K; k++) s += a[k] * w[k];
    if (act == 1) s = fmaxf(s, 0.0f);
    else if (act == 2) s = expf(s);
    out[(size_t)m * N + n] = s;
}

// ---------------------------------------------------------------------------
// Mixture sample z = where(choice < pi0, gaussian, gpd)
// ---------------------------------------------------------------------------
__global__ void z_kernel(const float* __restrict__ mean,
                         const float* __restrict__ logvar,
                         const float* __restrict__ scale,
                         const float* __restrict__ shape,
                         const float* __restrict__ eps,
                         const float* __restrict__ u,
                         const float* __restrict__ choice,
                         const float* __restrict__ pi,
                         float* __restrict__ z, int Bb, int Zd)
{
    int idx = blockIdx.x * blockDim.x + threadIdx.x;
    if (idx >= Bb * Zd) return;
    int b = idx / Zd;
    float p0 = pi[0], p1 = pi[1];
    float mx = fmaxf(p0, p1);
    float e0 = expf(p0 - mx), e1 = expf(p1 - mx);
    float pi0 = e0 / (e0 + e1);
    float zg = mean[idx] + eps[idx] * expf(0.5f * logvar[idx]);
    float sh = shape[idx];
    float zp = scale[idx] / sh * (powf(1.0f - u[idx], -sh) - 1.0f);
    z[idx] = (choice[b] < pi0) ? zg : zp;
}

// ---------------------------------------------------------------------------
// Host-side orchestration
// ---------------------------------------------------------------------------
extern "C" void kernel_launch(void* const* d_in, const int* in_sizes, int n_in,
                              void* d_out, int out_size, void* d_ws, size_t ws_size,
                              hipStream_t stream) {
    (void)in_sizes; (void)n_in; (void)out_size; (void)ws_size;
    const int Bb = 64, S = 512, DIN = 64, D = 512, L = 4, H = 8, Zd = 64,
              FUT = 96, FF = 2048;
    const int M = Bb * S; // 32768

    const float* X     = (const float*)d_in[0];
    const float* eps   = (const float*)d_in[1];
    const float* u     = (const float*)d_in[2];
    const float* choice= (const float*)d_in[3];
    const float* We    = (const float*)d_in[4];
    const float* be    = (const float*)d_in[5];
    const float* Wqkv  = (const float*)d_in[6];
    const float* bqkv  = (const float*)d_in[7];
    const float* Wo    = (const float*)d_in[8];
    const float* bo    = (const float*)d_in[9];
    const float* ln1g  = (const float*)d_in[10];
    const float* ln1b  = (const float*)d_in[11];
    const float* W1    = (const float*)d_in[12];
    const float* b1    = (const float*)d_in[13];
    const float* W2    = (const float*)d_in[14];
    const float* b2    = (const float*)d_in[15];
    const float* ln2g  = (const float*)d_in[16];
    const float* ln2b  = (const float*)d_in[17];
    const float* Wm    = (const float*)d_in[18];
    const float* bm    = (const float*)d_in[19];
    const float* Wlv   = (const float*)d_in[20];
    const float* blv   = (const float*)d_in[21];
    const float* Wsc   = (const float*)d_in[22];
    const float* bsc   = (const float*)d_in[23];
    const float* Wsh   = (const float*)d_in[24];
    const float* bsh   = (const float*)d_in[25];
    const float* pi_p  = (const float*)d_in[26];
    const float* Wfc1  = (const float*)d_in[27];
    const float* bfc1  = (const float*)d_in[28];
    const float* Wfc2  = (const float*)d_in[29];
    const float* bfc2  = (const float*)d_in[30];
    const float* Wout  = (const float*)d_in[31];
    const float* bout  = (const float*)d_in[32];
    const float* Wfin  = (const float*)d_in[33];
    const float* bfin  = (const float*)d_in[34];

    // ---- workspace layout (bytes) ----
    char* base = (char*)d_ws;
    const size_t szXF  = (size_t)M * D * 4;
    const size_t szXB  = (size_t)M * D * 2;
    const size_t szRA  = (size_t)M * (3 * D) * 2 + szXB; // qkv+o, reused as FF1
    const size_t szWB  = ((size_t)D * DIN + (size_t)L * (3 * D * D + D * D + FF * D + D * FF)) * 2;
    float* xf32         = (float*)(base);
    float* tmpf         = (float*)(base + szXF);
    unsigned short* xbf = (unsigned short*)(base + 2 * szXF);
    char*  rA           = base + 2 * szXF + szXB;
    unsigned short* qkvbf = (unsigned short*)rA;
    unsigned short* obf   = (unsigned short*)(rA + (size_t)M * 3 * D * 2);
    unsigned short* ff1bf = (unsigned short*)rA;   // lifetime-disjoint reuse
    unsigned short* wbf   = (unsigned short*)(rA + szRA);
    unsigned short* xinbf = (unsigned short*)((char*)wbf + szWB);
    float* zbuf  = (float*)((char*)xinbf + (size_t)M * DIN * 2);
    float* d1    = zbuf + Bb * Zd;
    float* d2    = d1 + Bb * 128;
    float* o96   = d2 + Bb * 500;

    // bf16 weight regions (element offsets inside wbf)
    const size_t oWe   = 0;
    const size_t oWqkv = oWe + (size_t)D * DIN;
    const size_t oWo   = oWqkv + (size_t)L * 3 * D * D;
    const size_t oW1   = oWo + (size_t)L * D * D;
    const size_t oW2   = oW1 + (size_t)L * FF * D;

    // ---- convert inputs / weights to bf16 once ----
    auto cvt = [&](const float* src, unsigned short* dst, size_t n) {
        f2bf_kernel<<<dim3((unsigned)((n + 255) / 256)), 256, 0, stream>>>(src, dst, (int)n);
    };
    cvt(X,    xinbf,       (size_t)M * DIN);
    cvt(We,   wbf + oWe,   (size_t)D * DIN);
    cvt(Wqkv, wbf + oWqkv, (size_t)L * 3 * D * D);
    cvt(Wo,   wbf + oWo,   (size_t)L * D * D);
    cvt(W1,   wbf + oW1,   (size_t)L * FF * D);
    cvt(W2,   wbf + oW2,   (size_t)L * D * FF);

    // ---- embedding: x = X @ We^T + be  (fp32 + bf16) ----
    gemm_wmma<<<dim3(D / 128, M / 128), 256, 0, stream>>>(
        xinbf, wbf + oWe, be, nullptr, xf32, xbf, M, D, DIN, 0);

    // ---- transformer layers ----
    for (int l = 0; l < L; l++) {
        // qkv
        gemm_wmma<<<dim3(3 * D / 128, M / 128), 256, 0, stream>>>(
            xbf, wbf + oWqkv + (size_t)l * 3 * D * D, bqkv + (size_t)l * 3 * D,
            nullptr, nullptr, qkvbf, M, 3 * D, D, 0);
        // attention
        attn_wmma<<<dim3(S / 64, H, Bb), 128, 0, stream>>>(qkvbf, obf, S, D, H);
        // Wo + residual
        gemm_wmma<<<dim3(D / 128, M / 128), 256, 0, stream>>>(
            obf, wbf + oWo + (size_t)l * D * D, bo + (size_t)l * D,
            xf32, tmpf, nullptr, M, D, D, 0);
        ln_kernel<<<M, 128, 0, stream>>>(tmpf, ln1g + (size_t)l * D,
                                         ln1b + (size_t)l * D, xf32, xbf, D);
        // FF1 (relu)
        gemm_wmma<<<dim3(FF / 128, M / 128), 256, 0, stream>>>(
            xbf, wbf + oW1 + (size_t)l * FF * D, b1 + (size_t)l * FF,
            nullptr, nullptr, ff1bf, M, FF, D, 1);
        // FF2 + residual
        gemm_wmma<<<dim3(D / 128, M / 128), 256, 0, stream>>>(
            ff1bf, wbf + oW2 + (size_t)l * D * FF, b2 + (size_t)l * D,
            xf32, tmpf, nullptr, M, D, FF, 0);
        ln_kernel<<<M, 128, 0, stream>>>(tmpf, ln2g + (size_t)l * D,
                                         ln2b + (size_t)l * D, xf32, xbf, D);
    }

    // ---- head ----
    float* outD   = (float*)d_out;
    float* recO   = outD;                // [64,96]
    float* meanO  = outD + Bb * FUT;     // [64,64]
    float* lvO    = meanO + Bb * Zd;
    float* scO    = lvO + Bb * Zd;
    float* shO    = scO + Bb * Zd;
    const float* hptr = xf32 + (size_t)(S - 1) * D;  // row stride S*D

    auto sg = [&](const float* A, int lda, const float* Wt, const float* bs,
                  float* out, int m, int n, int k, int act) {
        small_gemm<<<dim3((unsigned)((m * n + 255) / 256)), 256, 0, stream>>>(
            A, lda, Wt, bs, out, m, n, k, act);
    };
    sg(hptr, S * D, Wm,  bm,  meanO, Bb, Zd, D, 0);
    sg(hptr, S * D, Wlv, blv, lvO,   Bb, Zd, D, 0);
    sg(hptr, S * D, Wsc, bsc, scO,   Bb, Zd, D, 2);  // exp
    sg(hptr, S * D, Wsh, bsh, shO,   Bb, Zd, D, 0);

    z_kernel<<<dim3((Bb * Zd + 255) / 256), 256, 0, stream>>>(
        meanO, lvO, scO, shO, eps, u, choice, pi_p, zbuf, Bb, Zd);

    sg(zbuf, Zd,  Wfc1, bfc1, d1,   Bb, 128, Zd,  1);
    sg(d1,   128, Wfc2, bfc2, d2,   Bb, 500, 128, 1);
    sg(d2,   500, Wout, bout, o96,  Bb, FUT, 500, 0);
    sg(o96,  FUT, Wfin, bfin, recO, Bb, FUT, FUT, 0);
}
